// GIE_8675833938144
// MI455X (gfx1250) — compile-verified, matrix operand order checked by reading, not run on hardware
//
#include <hip/hip_runtime.h>

#define N_ENT 40943
#define RANK 500
#define BATCH 1024
#define NBR 5
#define INPUT_SIZE 600
#define KPAD 1024
#define KREAL (2*RANK)          // 1000
#define SCALE 0.044721359549995794f   // 1/sqrt(500)
#define MIN_NORM_F 1e-15f
#define BALL_EPS 0.004f

typedef __attribute__((ext_vector_type(16))) __bf16 v16bf;
typedef __attribute__((ext_vector_type(8)))  float  v8f;

__device__ __forceinline__ float softplusf(float x) {
  if (x > 20.f) return x;
  if (x < -20.f) return expf(x);
  return log1pf(expf(x));
}
__device__ __forceinline__ float artanhf_(float x) {
  x = fminf(fmaxf(x, -1.f + 1e-5f), 1.f - 1e-5f);
  return 0.5f * (log1pf(x) - log1pf(-x));
}

__device__ __forceinline__ float blk_sum(float v, float* red) {
  const int t = threadIdx.x;
  red[t] = v;
  __syncthreads();
  for (int s = 128; s > 0; s >>= 1) {
    if (t < s) red[t] += red[t + s];
    __syncthreads();
  }
  float r = red[0];
  __syncthreads();
  return r;
}

// ---------------------------------------------------------------------------
// Kernel 1: one block (256 threads) per batch row. Produces packed bf16 row
// X[i, 0:500]=A, X[i, 500:1000]=B, X[i, 1000:1024]=0 in workspace.
// ---------------------------------------------------------------------------
__global__ __launch_bounds__(256) void gie_prep(
    const int* __restrict__ x, const int* __restrict__ neighbors,
    const float* __restrict__ position, const float* __restrict__ emb_e,
    const float* __restrict__ emb_r, const float* __restrict__ emb1_r,
    const float* __restrict__ context_vec, const float* __restrict__ neighbor_vec_w,
    const float* __restrict__ position_vec_w,
    const float* __restrict__ c1_p, const float* __restrict__ c2_p,
    const float* __restrict__ ab_p,
    const float* __restrict__ fc1_w, const float* __restrict__ fc1_b,
    const float* __restrict__ fc2_w, const float* __restrict__ fc2_b,
    const float* __restrict__ fc3_w, const float* __restrict__ fc3_b,
    const float* __restrict__ ln_g, const float* __restrict__ ln_b,
    const float* __restrict__ fc_mn_w, const float* __restrict__ fc_mn_b,
    __bf16* __restrict__ Xb)
{
  __shared__ float s_in[608];   // npos input (600) -> fc2 out / LN buffer (500)
  __shared__ float sA[512];     // att_n -> lhs
  __shared__ float sB[512];     // fc1 out -> fc3 out (np)
  __shared__ float sC[512];     // head1
  __shared__ float sD[512];     // rot(head1)
  __shared__ float sE[512];     // rotation cos/sin cache
  __shared__ float red[256];
  __shared__ int   s_nbe[NBR], s_nbr[NBR];

  const int i = blockIdx.x;
  const int t = threadIdx.x;
  const int h = x[i*3 + 0];
  const int r = x[i*3 + 1];

  if (t < NBR) {
    s_nbe[t] = neighbors[(size_t)h*NBR*2 + t*2 + 0];
    s_nbr[t] = neighbors[(size_t)h*NBR*2 + t*2 + 1];
  }
  __syncthreads();

  // ---- neighbor attention ----
  float wlog[NBR];
  for (int j = 0; j < NBR; ++j) {
    const float* nvp = neighbor_vec_w + (size_t)s_nbr[j]*RANK;
    const float* nrp = emb_r + (size_t)s_nbr[j]*KREAL;       // first RANK cols
    float p = 0.f;
    for (int k = t; k < RANK; k += 256) p += nvp[k]*nrp[k];
    wlog[j] = blk_sum(p, red) * SCALE;
  }
  float mx = wlog[0];
  for (int j = 1; j < NBR; ++j) mx = fmaxf(mx, wlog[j]);
  float se = 0.f, wj[NBR];
  for (int j = 0; j < NBR; ++j) { wj[j] = expf(wlog[j]-mx); se += wj[j]; }
  for (int j = 0; j < NBR; ++j) wj[j] /= se;

  for (int k = t; k < RANK; k += 256) {
    float a = 0.f;
    for (int j = 0; j < NBR; ++j) a += wj[j] * emb_e[(size_t)s_nbe[j]*KREAL + k];
    sA[k] = a;
  }

  float ab = 0.f;
  for (int j = 0; j < NBR; ++j) ab += softplusf(ab_p[s_nbr[j]]);
  ab *= (1.f/NBR);

  // ---- position MLP ----
  for (int k = t; k < INPUT_SIZE; k += 256)
    s_in[k] = position[(size_t)h*INPUT_SIZE + k] * position_vec_w[(size_t)h*INPUT_SIZE + k] * SCALE;
  __syncthreads();

  for (int o = t; o < RANK; o += 256) {                      // fc1
    const float* wrow = fc1_w + (size_t)o*INPUT_SIZE;
    float acc = fc1_b[o];
    for (int k = 0; k < INPUT_SIZE; ++k) acc += s_in[k]*wrow[k];
    sB[o] = acc;
  }
  __syncthreads();
  for (int o = t; o < RANK; o += 256) {                      // fc2
    const float* wrow = fc2_w + (size_t)o*RANK;
    float acc = fc2_b[o];
    for (int k = 0; k < RANK; ++k) acc += sB[k]*wrow[k];
    s_in[o] = acc;
  }
  __syncthreads();
  // layernorm
  float ps = 0.f, ps2 = 0.f;
  for (int k = t; k < RANK; k += 256) { float v = s_in[k]; ps += v; ps2 += v*v; }
  float sum  = blk_sum(ps,  red);
  float sum2 = blk_sum(ps2, red);
  float mu  = sum / (float)RANK;
  float var = sum2 / (float)RANK - mu*mu;
  float inv = rsqrtf(var + 1e-5f);
  for (int k = t; k < RANK; k += 256)
    s_in[k] = (s_in[k]-mu)*inv*ln_g[k] + ln_b[k];
  __syncthreads();
  for (int o = t; o < RANK; o += 256) {                      // fc3
    const float* wrow = fc3_w + (size_t)o*RANK;
    float acc = fc3_b[o];
    for (int k = 0; k < RANK; ++k) acc += s_in[k]*wrow[k];
    sB[o] = acc;
  }
  __syncthreads();

  const float mn = softplusf(fc_mn_w[r] + fc_mn_b[0]);

  // ---- lhs ----
  const float* lhsf = emb_e + (size_t)h*KREAL;
  for (int k = t; k < RANK; k += 256)
    sA[k] = lhsf[k] + ab*sA[k] + mn*sB[k];
  __syncthreads();

  const float c1 = softplusf(c1_p[r]);
  const float c2 = softplusf(c2_p[r]);
  const float sq1 = sqrtf(c1), sq2 = sqrtf(c2);

  // head1 = expmap0(lhs, c1) = s1 * lhs (scalar multiple)
  float p = 0.f;
  for (int k = t; k < RANK; k += 256) { float v = sA[k]; p += v*v; }
  float nl = fmaxf(sqrtf(blk_sum(p, red)), MIN_NORM_F);
  float g1 = tanhf(sq1*nl)/(sq1*nl);
  float ng = g1*nl;
  float maxn1 = (1.f-BALL_EPS)/sq1;
  float s1 = g1 * ((ng > maxn1) ? (maxn1/ng) : 1.f);
  for (int k = t; k < RANK; k += 256) sC[k] = s1*sA[k];
  __syncthreads();

  // Givens rotation coefficients (normalized rel2 pairs)
  const float* rel2 = emb1_r + (size_t)r*KREAL;
  for (int pi = t; pi < RANK/2; pi += 256) {
    float gr = rel2[2*pi], gi = rel2[2*pi+1];
    float gn = fmaxf(sqrtf(gr*gr + gi*gi), MIN_NORM_F);
    sE[2*pi]   = gr/gn;
    sE[2*pi+1] = gi/gn;
  }
  __syncthreads();
  // D = rot(head1); all three candidates are scalar multiples of D
  for (int pi = t; pi < RANK/2; pi += 256) {
    float cc = sE[2*pi], ss = sE[2*pi+1];
    float x0 = sC[2*pi], x1 = sC[2*pi+1];
    sD[2*pi]   = cc*x0 - ss*x1;
    sD[2*pi+1] = cc*x1 + ss*x0;
  }
  __syncthreads();

  float pd = 0.f;
  for (int k = t; k < RANK; k += 256) { float v = sD[k]; pd += v*v; }
  float nD = fmaxf(sqrtf(blk_sum(pd, red)), MIN_NORM_F);
  float alpha1 = artanhf_(sq1*nD)/(sq1*nD);                  // res_c1 = alpha1*D

  float n1 = fmaxf(s1*nl, MIN_NORM_F);                       // |head1|
  float g2 = tanhf(sq2*n1)/(sq2*n1);
  float ng2 = g2*n1;
  float maxn2 = (1.f-BALL_EPS)/sq2;
  float s2 = g2 * ((ng2 > maxn2) ? (maxn2/ng2) : 1.f);       // head2 = s2*head1
  float n2d = fmaxf(s2*nD, MIN_NORM_F);
  float alpha2 = artanhf_(sq2*n2d)/(sq2*n2d);                // res_c2 = alpha2*s2*D

  // candidate attention collapses to scalar q
  const float* cvp = context_vec + (size_t)r*RANK;
  float pq = 0.f;
  for (int k = t; k < RANK; k += 256) pq += cvp[k]*sD[k];
  float d0 = blk_sum(pq, red) * SCALE;
  float L0 = alpha1*d0, L1 = alpha2*s2*d0, L2 = d0;
  float mq = fmaxf(L0, fmaxf(L1, L2));
  float e0 = expf(L0-mq), e1 = expf(L1-mq), e2 = expf(L2-mq);
  float q = (e0*alpha1 + e1*alpha2*s2 + e2) / (e0+e1+e2);    // att_q = q*D

  // pack A|B row as bf16 (padded to KPAD)
  const float* relf = emb_r + (size_t)r*KREAL;
  __bf16* xrow = Xb + (size_t)i*KPAD;
  for (int k = t; k < RANK; k += 256) {
    float rel0 = relf[k], relb = relf[RANK+k], l1v = lhsf[RANK+k];
    float aq = q*sD[k];
    xrow[k]        = (__bf16)(aq*rel0 - l1v*relb);
    xrow[RANK + k] = (__bf16)(aq*relb + l1v*rel0);
  }
  for (int k = KREAL + t; k < KPAD; k += 256) xrow[k] = (__bf16)0.f;
}

// ---------------------------------------------------------------------------
// Kernel 2: scores(1024 x 40943) = X(1024 x 1000) @ emb_e^T, bf16 WMMA.
// Block = 256 threads (8 waves) -> 64x128 output tile; wave -> 16x64 strip,
// 4 WMMAs per k-step with A-fragment reuse. X tile staged via
// GLOBAL_LOAD_ASYNC_TO_LDS_B128 (ASYNCcnt), W tile converted f32->bf16.
// ---------------------------------------------------------------------------
__global__ __launch_bounds__(256) void gie_gemm(
    const __bf16* __restrict__ Xb, const float* __restrict__ emb_e,
    float* __restrict__ out)
{
  __shared__ __align__(32) __bf16 As[64*32];    // M x K tile  (4 KB)
  __shared__ __align__(32) __bf16 Bs[128*32];   // N x K tile  (8 KB)

  const int tid  = threadIdx.x;
  const int lane = tid & 31;
  const int wave = tid >> 5;
  const int wm   = wave & 3;    // 4 M sub-tiles of 16
  const int wn   = wave >> 2;   // 2 N strips of 64
  const int m0   = blockIdx.y * 64;
  const int n0   = blockIdx.x * 128;

  const int sr = tid >> 2;      // staging row 0..63
  const int sc = tid & 3;       // 8-element segment 0..3

  // LDS byte address for this thread's async destination (wave-relative
  // offset = low 32 bits of the flat shared address, per aperture mapping)
  const unsigned int lA =
      (unsigned int)(unsigned long long)(uintptr_t)(&As[sr*32 + sc*8]);

  v8f acc[4] = { {}, {}, {}, {} };

  for (int kt = 0; kt < KPAD/32; ++kt) {
    const int kb = kt*32 + sc*8;

    // ---- stage X tile: async global -> LDS, 16B per lane, no VGPR bounce ----
    const unsigned long long gx =
        (unsigned long long)(uintptr_t)(Xb + (size_t)(m0+sr)*KPAD + kb);
    asm volatile("global_load_async_to_lds_b128 %0, %1, off"
                 :: "v"(lA), "v"(gx) : "memory");

    // ---- stage W tile: emb_e rows (N-major), f32 -> bf16, 2 rows/thread ----
    #pragma unroll
    for (int half = 0; half < 2; ++half) {
      const int jrow = sr + half*64;
      const int j = n0 + jrow;
      float wv[8];
      if (j < N_ENT && kb < KREAL) {
        const float* wp = emb_e + (size_t)j*KREAL + kb;
        const float4 f0 = *reinterpret_cast<const float4*>(wp);
        const float4 f1 = *reinterpret_cast<const float4*>(wp + 4);
        wv[0]=f0.x; wv[1]=f0.y; wv[2]=f0.z; wv[3]=f0.w;
        wv[4]=f1.x; wv[5]=f1.y; wv[6]=f1.z; wv[7]=f1.w;
        __builtin_prefetch(wp + 32, 0, 0);   // next k-block of this row
      } else {
        #pragma unroll
        for (int u = 0; u < 8; ++u) wv[u] = 0.f;
      }
      #pragma unroll
      for (int u = 0; u < 8; ++u) Bs[jrow*32 + sc*8 + u] = (__bf16)wv[u];
    }

    // own async copy must land before we signal the barrier
    asm volatile("s_wait_asynccnt 0x0" ::: "memory");
    __syncthreads();

    // ---- fragments: lanes 0-15 K=0..15, lanes 16-31 K=16..31 ----
    const int kh   = (lane >> 4) * 16;
    const int rsel = lane & 15;
    const v16bf a = *reinterpret_cast<const v16bf*>(&As[(wm*16 + rsel)*32 + kh]);
    #pragma unroll
    for (int c = 0; c < 4; ++c) {
      const v16bf b =
          *reinterpret_cast<const v16bf*>(&Bs[(wn*64 + c*16 + rsel)*32 + kh]);
      acc[c] = __builtin_amdgcn_wmma_f32_16x16x32_bf16(false, a, false, b,
                                                       (short)0, acc[c],
                                                       false, false);
    }
    __syncthreads();
  }

  // C/D layout: VGPR v -> M = v + 8*(lane>=16); N = lane&15
  const int row  = m0 + wm*16 + (lane >> 4)*8;
  const int colb = n0 + wn*64 + (lane & 15);
  #pragma unroll
  for (int c = 0; c < 4; ++c) {
    const int col = colb + c*16;
    if (col < N_ENT) {
      #pragma unroll
      for (int v = 0; v < 8; ++v)
        out[(size_t)(row + v)*N_ENT + col] = acc[c][v];
    }
  }
}

extern "C" void kernel_launch(void* const* d_in, const int* in_sizes, int n_in,
                              void* d_out, int out_size, void* d_ws, size_t ws_size,
                              hipStream_t stream) {
  const int*   x              = (const int*)d_in[0];
  const int*   neighbors      = (const int*)d_in[1];
  const float* position       = (const float*)d_in[2];
  const float* emb_e          = (const float*)d_in[3];
  const float* emb_r          = (const float*)d_in[4];
  const float* emb1_r         = (const float*)d_in[5];
  const float* context_vec    = (const float*)d_in[6];
  const float* neighbor_vec_w = (const float*)d_in[7];
  const float* position_vec_w = (const float*)d_in[8];
  const float* c1_p           = (const float*)d_in[9];
  const float* c2_p           = (const float*)d_in[10];
  const float* ab_p           = (const float*)d_in[11];
  const float* fc1_w          = (const float*)d_in[12];
  const float* fc1_b          = (const float*)d_in[13];
  const float* fc2_w          = (const float*)d_in[14];
  const float* fc2_b          = (const float*)d_in[15];
  const float* fc3_w          = (const float*)d_in[16];
  const float* fc3_b          = (const float*)d_in[17];
  const float* ln_g           = (const float*)d_in[18];
  const float* ln_b           = (const float*)d_in[19];
  const float* fc_mn_w        = (const float*)d_in[20];
  const float* fc_mn_b        = (const float*)d_in[21];

  __bf16* Xb = (__bf16*)d_ws;   // BATCH*KPAD bf16 = 2 MB

  gie_prep<<<BATCH, 256, 0, stream>>>(
      x, neighbors, position, emb_e, emb_r, emb1_r, context_vec,
      neighbor_vec_w, position_vec_w, c1_p, c2_p, ab_p,
      fc1_w, fc1_b, fc2_w, fc2_b, fc3_w, fc3_b, ln_g, ln_b,
      fc_mn_w, fc_mn_b, Xb);

  dim3 grid((N_ENT + 127)/128, BATCH/64);
  gie_gemm<<<grid, 256, 0, stream>>>(Xb, emb_e, (float*)d_out);
}
